// SE2MultiSelfAttention_7310034337915
// MI455X (gfx1250) — matrix operand order repaired and checked by reading, not compiled, not run on hardware
//
#include <hip/hip_runtime.h>
#include <math.h>

// ---------------------------------------------------------------------------
// SE2 multi self-attention, MI455X (gfx1250, wave32, WMMA bf16 16x16x32).
// Complex GEMMs run as 4 real v_wmma_f32_16x16x32_bf16 per K-step into two
// f32 accumulators; negation folded into f32->bf16 fragment conversion.
// Each wave computes a 16x32 complex output strip (two 16x16 tiles sharing
// the A fragments) to double WMMA issue per fragment conversion.
// ---------------------------------------------------------------------------

typedef __attribute__((ext_vector_type(16))) __bf16 v16bf;
typedef __attribute__((ext_vector_type(8)))  float  v8f;

#define B_    4
#define MAXM_ 4
#define DIM_  512
#define N_    1024
#define NH_   8
#define QD_   64

// E plane offset: layout E[e][b][k][512][1024]
__device__ __forceinline__ long EP(int e, int b, int k) {
  return ((long)((e * B_ + b) * MAXM_ + k)) * (long)(DIM_ * N_);
}

__device__ __forceinline__ v8f wmma_bf16(v16bf a, v16bf b, v8f c) {
  return __builtin_amdgcn_wmma_f32_16x16x32_bf16(false, a, false, b, (short)0, c,
                                                 false, false);
}

__device__ __forceinline__ void pf(const float* p) { __builtin_prefetch(p, 0, 1); }

// A fragment 16x32 bf16. Source row-major, p_ -> [row0][kk], row stride lda.
// Lane L: m=L%16, h=L/16; a[j]=A[m][h*8+j], a[8+j]=A[m][16+h*8+j].
__device__ __forceinline__ v16bf frag_a(const float* __restrict__ p_, long lda, int lane) {
  const int m = lane & 15, h = lane >> 4;
  const float* p = p_ + (long)m * lda + h * 8;
  v16bf a;
#pragma unroll
  for (int j = 0; j < 8; ++j) { a[j] = (__bf16)p[j]; a[8 + j] = (__bf16)p[16 + j]; }
  return a;
}

// Same, producing both +A and -A fragments (for the complex cross term).
__device__ __forceinline__ void frag_a_pairneg(const float* __restrict__ p_, long lda,
                                               int lane, v16bf& a, v16bf& ma) {
  const int m = lane & 15, h = lane >> 4;
  const float* p = p_ + (long)m * lda + h * 8;
#pragma unroll
  for (int j = 0; j < 8; ++j) {
    float x = p[j], y = p[16 + j];
    a[j]     = (__bf16)x;  ma[j]     = (__bf16)(-x);
    a[8 + j] = (__bf16)y;  ma[8 + j] = (__bf16)(-y);
  }
}

// A^T fragment: logical A[m][kk'] = src[kk'][m]; p_ -> [kk0][col0], row stride lds.
__device__ __forceinline__ v16bf frag_aT(const float* __restrict__ p_, long lds, int lane) {
  const int m = lane & 15, h = lane >> 4;
  const float* p = p_ + (long)(h * 8) * lds + m;
  v16bf a;
#pragma unroll
  for (int j = 0; j < 8; ++j) {
    a[j]     = (__bf16)p[(long)j * lds];
    a[8 + j] = (__bf16)p[(long)(16 + j) * lds];
  }
  return a;
}

// B fragment 32x16 bf16. Source row-major KxN, p_ -> [kk][col0], row stride ldb.
// Lane L: n=L%16, h=L/16; b[j]=B[h*8+j][n], b[8+j]=B[16+h*8+j][n].
__device__ __forceinline__ v16bf frag_b(const float* __restrict__ p_, long ldb, int lane) {
  const int n = lane & 15, h = lane >> 4;
  const float* p = p_ + (long)(h * 8) * ldb + n;
  v16bf b;
#pragma unroll
  for (int j = 0; j < 16; ++j) b[j] = (__bf16)p[(long)(j + (j & 8)) * ldb];
  return b;
}

// B fragment triple for scores: Q2 = conj(E0) * soft; emits Re, Im, -Im.
__device__ __forceinline__ void frag_b_conj_scaled(const float* __restrict__ pr_,
                                                   const float* __restrict__ pi_,
                                                   long ldb, int lane, float sr, float si,
                                                   v16bf& br, v16bf& bi, v16bf& mbi) {
  const int n = lane & 15, h = lane >> 4;
  const long off = (long)(h * 8) * ldb + n;
#pragma unroll
  for (int j = 0; j < 16; ++j) {
    float er = pr_[off + (long)(j + (j & 8)) * ldb];
    float ei = pi_[off + (long)(j + (j & 8)) * ldb];
    float qr = er * sr + ei * si;   // Re{(er - i ei)(sr + i si)}
    float qi = er * si - ei * sr;   // Im{...}
    br[j] = (__bf16)qr; bi[j] = (__bf16)qi; mbi[j] = (__bf16)(-qi);
  }
}

// C/D f32 16x16 store: lane L holds rows h*8+r, column n=L%16.
__device__ __forceinline__ void store_tile(float* __restrict__ C, long ldc, int row0,
                                           int col0, v8f acc, int lane) {
  const int n = lane & 15, h = lane >> 4;
  float* p = C + (long)(row0 + h * 8) * ldc + col0 + n;
#pragma unroll
  for (int r = 0; r < 8; ++r) p[(long)r * ldc] = acc[r];
}

// ---------------------------------------------------------------------------
// Kernel 1: E[e,b,k] = emb[e,k] @ x[b,k]   (complex 512x512x1024)
// One wave per 16x32 complex strip: tiles = 3*4*4*32*32 = 49152 (6144 blocks)
// ---------------------------------------------------------------------------
__global__ void k_embed_gemm(const float* __restrict__ embr, const float* __restrict__ embi,
                             const float* __restrict__ xr,   const float* __restrict__ xi,
                             float* __restrict__ Er, float* __restrict__ Ei) {
  int w = blockIdx.x * 8 + (threadIdx.x >> 5);
  const int lane = threadIdx.x & 31;
  const int nt = w & 31; w >>= 5;     // 32-wide N strips
  const int mt = w & 31; w >>= 5;
  const int k  = w & 3;  w >>= 2;
  const int b  = w & 3;  w >>= 2;
  const int e  = w;

  const float* Ar = embr + (long)(e * MAXM_ + k) * (DIM_ * DIM_) + (long)(mt * 16) * DIM_;
  const float* Ai = embi + (long)(e * MAXM_ + k) * (DIM_ * DIM_) + (long)(mt * 16) * DIM_;
  const float* Br = xr + (long)(b * MAXM_ + k) * (DIM_ * N_) + nt * 32;
  const float* Bi = xi + (long)(b * MAXM_ + k) * (DIM_ * N_) + nt * 32;

  v8f accr[2] = {}; v8f acci[2] = {};
  for (int kk = 0; kk < DIM_; kk += 32) {
    if (kk + 32 < DIM_) {                     // uniform branch; prefetch next step
      pf(Ar + (long)(lane & 15) * DIM_ + kk + 32);
      pf(Br + (long)(kk + 32 + lane) * N_);
    }
    v16bf ar = frag_a(Ar + kk, DIM_, lane);
    v16bf ai, mai; frag_a_pairneg(Ai + kk, DIM_, lane, ai, mai);
#pragma unroll
    for (int st = 0; st < 2; ++st) {
      v16bf br = frag_b(Br + (long)kk * N_ + st * 16, N_, lane);
      v16bf bi = frag_b(Bi + (long)kk * N_ + st * 16, N_, lane);
      accr[st] = wmma_bf16(ar,  br, accr[st]);
      accr[st] = wmma_bf16(mai, bi, accr[st]);   // - Ai*Bi
      acci[st] = wmma_bf16(ar,  bi, acci[st]);
      acci[st] = wmma_bf16(ai,  br, acci[st]);
    }
  }
  const long cb = EP(e, b, k);
#pragma unroll
  for (int st = 0; st < 2; ++st) {
    store_tile(Er + cb, N_, mt * 16, nt * 32 + st * 16, accr[st], lane);
    store_tile(Ei + cb, N_, mt * 16, nt * 32 + st * 16, acci[st], lane);
  }
}

// ---------------------------------------------------------------------------
// Kernel 2: scores A[b,h,i,j] = sum_{k,q} E1[p,i] * conj(E0)[p,j] * soft[k,h]
// A^T on the i side (strided); soft+conj folded into B fragments.
// One wave per 16x32 strip: tiles = 4*8*64*32 = 65536 (8192 blocks)
// ---------------------------------------------------------------------------
__global__ void k_scores(const float* __restrict__ Er, const float* __restrict__ Ei,
                         const float* __restrict__ softr, const float* __restrict__ softi,
                         float* __restrict__ Pr, float* __restrict__ Pi) {
  int w = blockIdx.x * 8 + (threadIdx.x >> 5);
  const int lane = threadIdx.x & 31;
  const int jt = w & 31; w >>= 5;     // 32-wide j strips
  const int it = w & 63; w >>= 6;
  const int h  = w & 7;  w >>= 3;
  const int b  = w;

  v8f accr[2] = {}; v8f acci[2] = {};
  for (int kk = 0; kk < MAXM_ * QD_; kk += 32) {
    const int k  = kk >> 6;          // 32-chunk always inside one harmonic
    const int q0 = kk & 63;
    const float sr = softr[k * NH_ + h], si = softi[k * NH_ + h];
    const long rowbase = (long)(h * QD_ + q0) * N_;
    const float* Kr_ = Er + EP(1, b, k) + rowbase;
    const float* Ki_ = Ei + EP(1, b, k) + rowbase;
    const float* Qr_ = Er + EP(0, b, k) + rowbase;
    const float* Qi_ = Ei + EP(0, b, k) + rowbase;

    v16bf akr = frag_aT(Kr_ + it * 16, N_, lane);
    v16bf aki = frag_aT(Ki_ + it * 16, N_, lane);
#pragma unroll
    for (int st = 0; st < 2; ++st) {
      v16bf qbr, qbi, mqbi;
      frag_b_conj_scaled(Qr_ + jt * 32 + st * 16, Qi_ + jt * 32 + st * 16, N_, lane,
                         sr, si, qbr, qbi, mqbi);
      accr[st] = wmma_bf16(akr, qbr,  accr[st]);
      accr[st] = wmma_bf16(aki, mqbi, accr[st]);  // - Ki*Q2i
      acci[st] = wmma_bf16(akr, qbi,  acci[st]);
      acci[st] = wmma_bf16(aki, qbr,  acci[st]);
    }
  }
  const long cb = (long)(b * NH_ + h) * N_ * N_;
#pragma unroll
  for (int st = 0; st < 2; ++st) {
    store_tile(Pr + cb, N_, it * 16, jt * 32 + st * 16, accr[st], lane);
    store_tile(Pi + cb, N_, it * 16, jt * 32 + st * 16, acci[st], lane);
  }
}

// ---------------------------------------------------------------------------
// Kernel 3: S[b,k,h,M] = soft[k,h] * sum_q conj(E0)[b,k,h*64+q,M] * enc[k,h,q]
// 131072 threads, 64-length complex reduction each.
// ---------------------------------------------------------------------------
__global__ void k_sreduce(const float* __restrict__ Er, const float* __restrict__ Ei,
                          const float* __restrict__ encr, const float* __restrict__ enci,
                          const float* __restrict__ softr, const float* __restrict__ softi,
                          float* __restrict__ Sr, float* __restrict__ Si) {
  const int t = blockIdx.x * blockDim.x + threadIdx.x;   // ((b*4+k)*8+h)*1024+M
  const int M = t & 1023;
  const int h = (t >> 10) & 7;
  const int k = (t >> 13) & 3;
  const int b = t >> 15;
  const float* qr = Er + EP(0, b, k) + (long)(h * QD_) * N_ + M;
  const float* qi = Ei + EP(0, b, k) + (long)(h * QD_) * N_ + M;
  float ar = 0.f, ai = 0.f;
  for (int q = 0; q < QD_; ++q) {
    float er = qr[(long)q * N_];
    float ei = -qi[(long)q * N_];                  // conj
    float cr = encr[(k * NH_ + h) * QD_ + q];
    float ci = enci[(k * NH_ + h) * QD_ + q];
    ar += er * cr - ei * ci;
    ai += er * ci + ei * cr;
  }
  const float sr = softr[k * NH_ + h], si = softi[k * NH_ + h];
  Sr[t] = ar * sr - ai * si;
  Si[t] = ar * si + ai * sr;
}

// ---------------------------------------------------------------------------
// Kernel 4: P[b,h,r,c] = |scores + sum_k S[b,k,h,r]*exp(i*k*(r-c)*2pi/N)| / 8
// written in place over the scores real plane.
// ---------------------------------------------------------------------------
__global__ void k_posabs(const float* __restrict__ Sr, const float* __restrict__ Si,
                         float* __restrict__ Pr, const float* __restrict__ Pi) {
  const long t = (long)blockIdx.x * blockDim.x + threadIdx.x; // ((b*8+h)*1024+r)*1024+c
  const int c = (int)(t & 1023);
  const int r = (int)((t >> 10) & 1023);
  const int h = (int)((t >> 20) & 7);
  const int b = (int)(t >> 23);
  const float th = (float)(r - c) * (6.28318530717958647692f / (float)N_);
  const float cw = cosf(th), sw = sinf(th);
  float zr = 1.f, zi = 0.f, pr = 0.f, pi = 0.f;
#pragma unroll
  for (int k = 0; k < MAXM_; ++k) {
    const long si_ = ((long)((b * MAXM_ + k) * NH_ + h)) * N_ + r;
    const float vr = Sr[si_], vi = Si[si_];
    pr += vr * zr - vi * zi;
    pi += vr * zi + vi * zr;
    const float nzr = zr * cw - zi * sw;
    zi = zr * sw + zi * cw;
    zr = nzr;
  }
  const float ur = Pr[t] + pr, ui = Pi[t] + pi;
  Pr[t] = sqrtf(ur * ur + ui * ui) * 0.125f;       // /sqrt(QD)
}

// ---------------------------------------------------------------------------
// Kernel 5: softmax over the row axis (axis -2) per (b,h,column), in place.
// 128 blocks x 256 threads: one thread per column.
// ---------------------------------------------------------------------------
__global__ void k_softmax(float* __restrict__ P) {
  const int bh = blockIdx.x >> 2;
  const int c  = ((blockIdx.x & 3) << 8) + threadIdx.x;
  float* col = P + (long)bh * N_ * N_ + c;
  float mx = -3.402823466e38f;
  for (int r = 0; r < N_; ++r) mx = fmaxf(mx, col[(long)r * N_]);
  float sum = 0.f;
  for (int r = 0; r < N_; ++r) sum += __expf(col[(long)r * N_] - mx);
  const float inv = 1.f / sum;
  for (int r = 0; r < N_; ++r) col[(long)r * N_] = __expf(col[(long)r * N_] - mx) * inv;
}

// ---------------------------------------------------------------------------
// Kernel 6: ctx[b,k,h,q,c] = sum_r V[b,k,h,q,r] * A_soft[b,h,r,c]
// complex x real: one wave per 16x32 strip: tiles = 4*8*16*32 = 16384 (2048 blocks)
// ---------------------------------------------------------------------------
__global__ void k_ctx(const float* __restrict__ Er, const float* __restrict__ Ei,
                      const float* __restrict__ P,
                      float* __restrict__ ctxr, float* __restrict__ ctxi) {
  int w = blockIdx.x * 8 + (threadIdx.x >> 5);
  const int lane = threadIdx.x & 31;
  const int nt = w & 31; w >>= 5;     // 32-wide c strips
  const int pt = w & 15; w >>= 4;
  const int h  = w & 7;  w >>= 3;
  const int b  = w;
  const int p0 = pt * 16;
  const int k  = p0 >> 6;
  const int q0 = p0 & 63;

  const float* Vr = Er + EP(2, b, k) + (long)(h * QD_ + q0) * N_;
  const float* Vi = Ei + EP(2, b, k) + (long)(h * QD_ + q0) * N_;
  const float* Bp = P + (long)(b * NH_ + h) * N_ * N_ + nt * 32;

  v8f accr[2] = {}; v8f acci[2] = {};
  for (int kk = 0; kk < N_; kk += 32) {
    if (kk + 32 < N_) {
      pf(Vr + (long)(lane & 15) * N_ + kk + 32);
      pf(Bp + (long)(kk + 32 + lane) * N_);
    }
    v16bf ar = frag_a(Vr + kk, N_, lane);
    v16bf ai = frag_a(Vi + kk, N_, lane);
#pragma unroll
    for (int st = 0; st < 2; ++st) {
      v16bf bb = frag_b(Bp + (long)kk * N_ + st * 16, N_, lane);
      accr[st] = wmma_bf16(ar, bb, accr[st]);
      acci[st] = wmma_bf16(ai, bb, acci[st]);
    }
  }
  const long row0 = (long)(b * MAXM_ + k) * DIM_ + h * QD_ + q0;
#pragma unroll
  for (int st = 0; st < 2; ++st) {
    store_tile(ctxr + row0 * N_, N_, 0, nt * 32 + st * 16, accr[st], lane);
    store_tile(ctxi + row0 * N_, N_, 0, nt * 32 + st * 16, acci[st], lane);
  }
}

// ---------------------------------------------------------------------------
// Kernel 7: res[b,k,d,c] = sum_c2 out[k,d,c2] * ctx[b,k,c2,c]  (complex)
// Output interleaved complex64 (re,im). tiles = 4*4*32*32 = 16384 (2048 blocks)
// ---------------------------------------------------------------------------
__global__ void k_res(const float* __restrict__ outr, const float* __restrict__ outi,
                      const float* __restrict__ ctxr, const float* __restrict__ ctxi,
                      float* __restrict__ R) {
  int w = blockIdx.x * 8 + (threadIdx.x >> 5);
  const int lane = threadIdx.x & 31;
  const int nt = w & 31; w >>= 5;     // 32-wide c strips
  const int mt = w & 31; w >>= 5;
  const int k  = w & 3;  w >>= 2;
  const int b  = w;

  const float* Ar = outr + (long)k * (DIM_ * DIM_) + (long)(mt * 16) * DIM_;
  const float* Ai = outi + (long)k * (DIM_ * DIM_) + (long)(mt * 16) * DIM_;
  const float* Br = ctxr + (long)(b * MAXM_ + k) * (DIM_ * N_) + nt * 32;
  const float* Bi = ctxi + (long)(b * MAXM_ + k) * (DIM_ * N_) + nt * 32;

  v8f accr[2] = {}; v8f acci[2] = {};
  for (int kk = 0; kk < DIM_; kk += 32) {
    if (kk + 32 < DIM_) {
      pf(Ar + (long)(lane & 15) * DIM_ + kk + 32);
      pf(Br + (long)(kk + 32 + lane) * N_);
    }
    v16bf ar = frag_a(Ar + kk, DIM_, lane);
    v16bf ai, mai; frag_a_pairneg(Ai + kk, DIM_, lane, ai, mai);
#pragma unroll
    for (int st = 0; st < 2; ++st) {
      v16bf br = frag_b(Br + (long)kk * N_ + st * 16, N_, lane);
      v16bf bi = frag_b(Bi + (long)kk * N_ + st * 16, N_, lane);
      accr[st] = wmma_bf16(ar,  br, accr[st]);
      accr[st] = wmma_bf16(mai, bi, accr[st]);
      acci[st] = wmma_bf16(ar,  bi, acci[st]);
      acci[st] = wmma_bf16(ai,  br, acci[st]);
    }
  }
  const int n = lane & 15, hh = lane >> 4;
#pragma unroll
  for (int st = 0; st < 2; ++st) {
    const long base =
        (((long)(b * MAXM_ + k) * DIM_) + mt * 16 + hh * 8) * N_ + nt * 32 + st * 16 + n;
#pragma unroll
    for (int r = 0; r < 8; ++r) {
      const long idx = base + (long)r * N_;
      R[2 * idx]     = accr[st][r];
      R[2 * idx + 1] = acci[st][r];
    }
  }
}

// ---------------------------------------------------------------------------
extern "C" void kernel_launch(void* const* d_in, const int* in_sizes, int n_in,
                              void* d_out, int out_size, void* d_ws, size_t ws_size,
                              hipStream_t stream) {
  (void)in_sizes; (void)n_in; (void)out_size; (void)ws_size;
  const float* xr    = (const float*)d_in[0];
  const float* xi    = (const float*)d_in[1];
  const float* embr  = (const float*)d_in[2];
  const float* embi  = (const float*)d_in[3];
  const float* encr  = (const float*)d_in[4];
  const float* enci  = (const float*)d_in[5];
  const float* softr = (const float*)d_in[6];
  const float* softi = (const float*)d_in[7];
  const float* outr  = (const float*)d_in[8];
  const float* outi  = (const float*)d_in[9];
  float* R  = (float*)d_out;
  float* ws = (float*)d_ws;

  const long E_PLANE = 3L * B_ * MAXM_ * DIM_ * N_;   // 25,165,824 floats
  const long S_ELEMS = (long)B_ * MAXM_ * NH_ * N_;   // 131,072
  const long P_PLANE = (long)B_ * NH_ * N_ * N_;      // 33,554,432

  float* Er = ws;
  float* Ei = Er + E_PLANE;
  float* Sr = Ei + E_PLANE;
  float* Si = Sr + S_ELEMS;
  float* Pr = Si + S_ELEMS;
  float* Pi = Pr + P_PLANE;
  float* ctxr = Pi;                                   // Pi reused after k_posabs
  float* ctxi = Pi + (long)B_ * MAXM_ * DIM_ * N_;    // 8,388,608 floats each

  k_embed_gemm<<< 6144, 256, 0, stream>>>(embr, embi, xr, xi, Er, Ei);
  k_scores    <<< 8192, 256, 0, stream>>>(Er, Ei, softr, softi, Pr, Pi);
  k_sreduce   <<<  512, 256, 0, stream>>>(Er, Ei, encr, enci, softr, softi, Sr, Si);
  k_posabs    <<<131072, 256, 0, stream>>>(Sr, Si, Pr, Pi);
  k_softmax   <<<  128, 256, 0, stream>>>(Pr);
  k_ctx       <<< 2048, 256, 0, stream>>>(Er, Ei, Pr, ctxr, ctxi);
  k_res       <<< 2048, 256, 0, stream>>>(outr, outi, ctxr, ctxi, R);
}
